// LateralLayerToy_6073083757245
// MI455X (gfx1250) — matrix operand order repaired and checked by reading, not compiled, not run on hardware
//
#include <hip/hip_runtime.h>
#include <math.h>

// ---- CDNA5 vector types ----
typedef __attribute__((ext_vector_type(16))) __bf16 v16bf;
typedef __attribute__((ext_vector_type(8)))  __bf16 v8bf;
typedef __attribute__((ext_vector_type(8)))  float  v8f;

#define BB   128
#define CC   32
#define HH   32
#define WW2  32
#define PS   11
#define LPAD 5
#define KWIN 10
#define HP   42           // padded spatial extent (32 + 2*5)
#define ALPHA 0.9f
#define LR    0.01f

// =====================================================================
// K1: weight prep.  wt = tanh(W); z[o] = sum_{i,p,q} wt;
// store bf16 hi/lo in B-operand-friendly layout  wb[(p*11+q)*32 + o][i]
// (lane = column o, K = i contiguous).
// =====================================================================
__global__ void k_wprep(const float* __restrict__ W,
                        __bf16* __restrict__ wb_hi, __bf16* __restrict__ wb_lo,
                        float* __restrict__ zbuf) {
  const int o = blockIdx.x;          // 32 blocks
  const int tid = threadIdx.x;       // 128 threads
  float zsum = 0.f;
  for (int e = tid; e < CC * PS * PS; e += blockDim.x) {
    const int i  = e / (PS * PS);
    const int pq = e % (PS * PS);
    const float t = tanhf(W[(o * CC + i) * PS * PS + pq]);
    zsum += t;
    const __bf16 hi = (__bf16)t;
    const __bf16 lo = (__bf16)(t - (float)hi);
    const int dst = (pq * CC + o) * CC + i;
    wb_hi[dst] = hi;
    wb_lo[dst] = lo;
  }
  __shared__ float red[128];
  red[tid] = zsum;
  __syncthreads();
  for (int s = 64; s > 0; s >>= 1) {
    if (tid < s) red[tid] += red[tid + s];
    __syncthreads();
  }
  if (tid == 0) zbuf[o] = red[0];
}

// =====================================================================
// K2: transpose + pad + bf16 split:  xT[b][hp][wp][i] (channels innermost)
// Zero-pads every call (deterministic, no stale ws state).
// =====================================================================
__global__ void k_xpose(const float* __restrict__ x,
                        __bf16* __restrict__ xt_hi, __bf16* __restrict__ xt_lo) {
  long long idx = (long long)blockIdx.x * blockDim.x + threadIdx.x;
  const long long total = (long long)BB * HP * HP * CC;
  if (idx >= total) return;
  const int i  = (int)(idx & 31);
  long long t  = idx >> 5;
  const int wp = (int)(t % HP); t /= HP;
  const int hp = (int)(t % HP); t /= HP;
  const int b  = (int)t;
  const int h = hp - LPAD, w = wp - LPAD;
  float v = 0.f;
  if ((unsigned)h < (unsigned)HH && (unsigned)w < (unsigned)WW2)
    v = x[((long long)(b * CC + i) * HH + h) * WW2 + w];
  const __bf16 hi = (__bf16)v;
  xt_hi[idx] = hi;
  xt_lo[idx] = (__bf16)(v - (float)hi);
}

// =====================================================================
// K3: implicit-GEMM conv via v_wmma_f32_16x16x32_bf16.
// One wave per output tile (b, h, 16 w-positions) x 32 output channels.
// K-loop: 121 (p,q) shifts, K=32 input channels per WMMA.
// hi/lo split: acc += Ah*Bh + Ah*Bl + Al*Bh  (fp32-class accuracy).
// Epilogue: p = a/(0.9 z[o]); out = p>0.5; p stashed for top-k.
// =====================================================================
__global__ void __launch_bounds__(128) k_conv(
    const __bf16* __restrict__ xt_hi, const __bf16* __restrict__ xt_lo,
    const __bf16* __restrict__ wb_hi, const __bf16* __restrict__ wb_lo,
    const float* __restrict__ zbuf,
    float* __restrict__ pbuf, float* __restrict__ out) {
  const int lane = threadIdx.x & 31;
  const int tile = blockIdx.x * 4 + (threadIdx.x >> 5);   // 0..8191
  const int b  = tile >> 6;
  const int h  = (tile >> 1) & 31;
  const int w0 = (tile & 1) << 4;

  const int m   = lane & 15;              // A row (spatial offset)
  const int kb  = (lane < 16) ? 0 : 16;   // A-operand K base (contiguous 16)
  const int bkb = (lane < 16) ? 0 : 16;   // B-operand K base (contiguous 16)

  v8f acc0 = {};
  v8f acc1 = {};

  const __bf16* xb_hi = xt_hi + (long long)b * HP * HP * CC;
  const __bf16* xb_lo = xt_lo + (long long)b * HP * HP * CC;

  for (int p = 0; p < PS; ++p) {
    const __bf16* xr_hi = xb_hi + (long long)(h + p) * HP * CC;
    const __bf16* xr_lo = xb_lo + (long long)(h + p) * HP * CC;
    for (int q = 0; q < PS; ++q) {
      // ---- A operand: row m = x[b][:, h+p-5, w0+m+q-5] (32 channels) ----
      const __bf16* ar_hi = xr_hi + (w0 + q + m) * CC;
      const __bf16* ar_lo = xr_lo + (w0 + q + m) * CC;
      const v8bf ah0 = *(const v8bf*)(ar_hi + kb);        // K = kb..kb+7
      const v8bf ah1 = *(const v8bf*)(ar_hi + kb + 8);    // K = kb+8..kb+15
      const v8bf al0 = *(const v8bf*)(ar_lo + kb);
      const v8bf al1 = *(const v8bf*)(ar_lo + kb + 8);
      v16bf a_hi, a_lo;
#pragma unroll
      for (int e = 0; e < 8; ++e) {
        a_hi[e] = ah0[e]; a_hi[8 + e] = ah1[e];
        a_lo[e] = al0[e]; a_lo[8 + e] = al1[e];
      }
      // ---- B operands: column o = weight[o][:, p, q] ----
      const int pq = p * PS + q;
      const __bf16* w0h = wb_hi + ((pq * CC) + (lane & 15)) * CC + bkb;
      const __bf16* w0l = wb_lo + ((pq * CC) + (lane & 15)) * CC + bkb;
      const v16bf b0h = *(const v16bf*)w0h;
      const v16bf b0l = *(const v16bf*)w0l;
      const v16bf b1h = *(const v16bf*)(w0h + 16 * CC);
      const v16bf b1l = *(const v16bf*)(w0l + 16 * CC);

      acc0 = __builtin_amdgcn_wmma_f32_16x16x32_bf16(false, a_hi, false, b0h,
                                                     (short)0, acc0, false, false);
      acc0 = __builtin_amdgcn_wmma_f32_16x16x32_bf16(false, a_hi, false, b0l,
                                                     (short)0, acc0, false, false);
      acc0 = __builtin_amdgcn_wmma_f32_16x16x32_bf16(false, a_lo, false, b0h,
                                                     (short)0, acc0, false, false);
      acc1 = __builtin_amdgcn_wmma_f32_16x16x32_bf16(false, a_hi, false, b1h,
                                                     (short)0, acc1, false, false);
      acc1 = __builtin_amdgcn_wmma_f32_16x16x32_bf16(false, a_hi, false, b1l,
                                                     (short)0, acc1, false, false);
      acc1 = __builtin_amdgcn_wmma_f32_16x16x32_bf16(false, a_lo, false, b1h,
                                                     (short)0, acc1, false, false);
    }
  }

  // ---- epilogue: D layout -> lane holds (M = j + (lane>=16)*8, N = lane&15)
  const int Moff = (lane >> 4) << 3;
#pragma unroll
  for (int ot = 0; ot < 2; ++ot) {
    const int o = ot * 16 + (lane & 15);
    const float inv = 1.f / (ALPHA * zbuf[o]);
    const v8f acc = ot ? acc1 : acc0;
    const long long base = ((long long)(b * CC + o) * HH + h) * WW2;
#pragma unroll
    for (int j = 0; j < 8; ++j) {
      const float pv = acc[j] * inv;
      const long long idx = base + (w0 + j + Moff);
      pbuf[idx] = pv;
      out[idx]  = (pv > 0.5f) ? 1.f : 0.f;
    }
  }
}

// =====================================================================
// K4: top-K=10 per (b,c) over 1024 p values.  One wave32 per (b,c).
// Iterative argmax with __shfl_xor reduction; tie -> lower flat index
// (matches jax top_k). Emits winner (h,w), x value, binary out value.
// =====================================================================
__global__ void __launch_bounds__(128) k_topk(
    const float* __restrict__ pbuf, const float* __restrict__ x,
    int* __restrict__ winH, int* __restrict__ winW,
    float* __restrict__ winX, float* __restrict__ winO) {
  const int lane = threadIdx.x & 31;
  const int bc   = blockIdx.x * 4 + (threadIdx.x >> 5);   // 0..4095
  const float* pp = pbuf + (long long)bc * 1024;

  float v[32];
#pragma unroll
  for (int j = 0; j < 32; ++j) v[j] = pp[j * 32 + lane];

  unsigned rm = 0u;
  for (int r = 0; r < KWIN; ++r) {
    float best = -3.4e38f;
    int bidx = 0x7fffffff;
#pragma unroll
    for (int j = 0; j < 32; ++j) {
      if (!((rm >> j) & 1u) && v[j] > best) { best = v[j]; bidx = j * 32 + lane; }
    }
#pragma unroll
    for (int s = 16; s > 0; s >>= 1) {
      const float ov = __shfl_xor(best, s, 32);
      const int   oi = __shfl_xor(bidx, s, 32);
      if (ov > best || (ov == best && oi < bidx)) { best = ov; bidx = oi; }
    }
    if ((bidx & 31) == lane) rm |= (1u << (bidx >> 5));
    if (lane == 0) {
      const int hh = bidx >> 5, ww = bidx & 31;
      const int slot = bc * KWIN + r;
      winH[slot] = hh;
      winW[slot] = ww;
      winX[slot] = x[(long long)bc * 1024 + hh * 32 + ww];
      winO[slot] = (best > 0.5f) ? 1.f : 0.f;
    }
  }
}

// =====================================================================
// K5: sparse Hebbian accumulation.  Per batch b:
//   Osum[o]      += out_val
//   Ssum[i,ps]   += x_val * cnt_w(z) * [y-ps+5 in range]
//   Aacc[i,o,ps] += x_val * out_val   for winner pairs with
//                   ps = y_i - h_o + 5 in [0,11), |z_i - w_o| <= 5
// =====================================================================
__global__ void __launch_bounds__(256) k_pairs(
    const int* __restrict__ winH, const int* __restrict__ winW,
    const float* __restrict__ winX, const float* __restrict__ winO,
    float* __restrict__ Aacc, float* __restrict__ Ssum, float* __restrict__ Osum) {
  __shared__ float lA[CC * CC * PS];   // 45056 B
  __shared__ float lS[CC * PS];
  __shared__ float lO[CC];
  __shared__ int   sh[CC * KWIN], sw[CC * KWIN];
  __shared__ float sx[CC * KWIN], so[CC * KWIN];

  const int b = blockIdx.x, tid = threadIdx.x;
  for (int e = tid; e < CC * CC * PS; e += 256) lA[e] = 0.f;
  for (int e = tid; e < CC * PS; e += 256) lS[e] = 0.f;
  if (tid < CC) lO[tid] = 0.f;
  for (int e = tid; e < CC * KWIN; e += 256) {
    const int g = b * CC * KWIN + e;
    sh[e] = winH[g]; sw[e] = winW[g]; sx[e] = winX[g]; so[e] = winO[g];
  }
  __syncthreads();

  // meanO + meanS contributions (each winner visited once)
  for (int ck = tid; ck < CC * KWIN; ck += 256) {
    const int c = ck / KWIN;
    const int y = sh[ck], z = sw[ck];
    const float xval = sx[ck];
    atomicAdd(&lO[c], so[ck]);
    const int qhi = (z + 5 < 10) ? z + 5 : 10;
    const int qlo = (z - 26 > 0) ? z - 26 : 0;
    const float xc = xval * (float)(qhi - qlo + 1);
    for (int ps = 0; ps < PS; ++ps) {
      const int hh = y - ps + 5;
      if (hh >= 0 && hh < HH) atomicAdd(&lS[c * PS + ps], xc);
    }
  }
  __syncthreads();

  // winner-pair accumulation for A
  for (int oj = tid; oj < CC * KWIN; oj += 256) {
    const float oval = so[oj];
    if (oval == 0.f) continue;
    const int o = oj / KWIN;
    const int ho = sh[oj], wo = sw[oj];
    for (int ik = 0; ik < CC * KWIN; ++ik) {
      const int ps = sh[ik] - ho + 5;
      const int dw = sw[ik] - wo;
      if ((unsigned)ps < (unsigned)PS && dw >= -5 && dw <= 5) {
        const int i = ik / KWIN;
        atomicAdd(&lA[(i * CC + o) * PS + ps], sx[ik] * oval);
      }
    }
  }
  __syncthreads();

  for (int e = tid; e < CC * CC * PS; e += 256)
    if (lA[e] != 0.f) atomicAdd(&Aacc[e], lA[e]);
  for (int e = tid; e < CC * PS; e += 256) atomicAdd(&Ssum[e], lS[e]);
  if (tid < CC) atomicAdd(&Osum[tid], lO[tid]);
}

// =====================================================================
// K6: finalize  new_W = W + LR * (3A - meanS - PS*meanO), broadcast over r.
// =====================================================================
__global__ void k_final(const float* __restrict__ W,
                        const float* __restrict__ Aacc,
                        const float* __restrict__ Ssum,
                        const float* __restrict__ Osum,
                        float* __restrict__ newW) {
  const int idx = blockIdx.x * blockDim.x + threadIdx.x;
  if (idx >= CC * CC * PS * PS) return;
  const int d0 = idx / (CC * PS * PS);
  const int r1 = idx % (CC * PS * PS);
  const int d1 = r1 / (PS * PS);
  const int pp = (r1 % (PS * PS)) / PS;
  const float invd = 1.f / (float)(BB * HH * WW2);
  const float Av = Aacc[(d0 * CC + d1) * PS + pp] * invd;
  const float Sv = Ssum[d0 * PS + pp] * invd;
  const float Ov = Osum[d1] * invd;
  newW[idx] = W[idx] + LR * (3.f * Av - Sv - (float)PS * Ov);
}

// =====================================================================
extern "C" void kernel_launch(void* const* d_in, const int* in_sizes, int n_in,
                              void* d_out, int out_size, void* d_ws, size_t ws_size,
                              hipStream_t stream) {
  (void)in_sizes; (void)n_in; (void)out_size; (void)ws_size;
  const float* x = (const float*)d_in[0];
  const float* W = (const float*)d_in[1];
  float* out  = (float*)d_out;                       // (128,32,32,32)
  float* newW = (float*)d_out + (size_t)BB * CC * HH * WW2;

  char* ws = (char*)d_ws;
  size_t off = 0;
  auto carve = [&](size_t bytes) -> void* {
    void* p = ws + off;
    off = (off + bytes + 255) & ~(size_t)255;
    return p;
  };

  const size_t XT_ELEMS = (size_t)BB * HP * HP * CC;       // 7,225,344
  __bf16* xt_hi = (__bf16*)carve(XT_ELEMS * 2);
  __bf16* xt_lo = (__bf16*)carve(XT_ELEMS * 2);
  __bf16* wb_hi = (__bf16*)carve((size_t)PS * PS * CC * CC * 2);
  __bf16* wb_lo = (__bf16*)carve((size_t)PS * PS * CC * CC * 2);
  float*  zbuf  = (float*)carve(CC * sizeof(float));
  float*  pbuf  = (float*)carve((size_t)BB * CC * HH * WW2 * sizeof(float));
  int*    winH  = (int*)carve((size_t)BB * CC * KWIN * sizeof(int));
  int*    winW  = (int*)carve((size_t)BB * CC * KWIN * sizeof(int));
  float*  winX  = (float*)carve((size_t)BB * CC * KWIN * sizeof(float));
  float*  winO  = (float*)carve((size_t)BB * CC * KWIN * sizeof(float));
  float*  Aacc  = (float*)carve((size_t)CC * CC * PS * sizeof(float));
  float*  Ssum  = (float*)carve((size_t)CC * PS * sizeof(float));
  float*  Osum  = (float*)carve((size_t)CC * sizeof(float));

  hipMemsetAsync(Aacc, 0, (size_t)CC * CC * PS * sizeof(float), stream);
  hipMemsetAsync(Ssum, 0, (size_t)CC * PS * sizeof(float), stream);
  hipMemsetAsync(Osum, 0, (size_t)CC * sizeof(float), stream);

  k_wprep<<<CC, 128, 0, stream>>>(W, wb_hi, wb_lo, zbuf);
  k_xpose<<<(int)(XT_ELEMS / 256), 256, 0, stream>>>(x, xt_hi, xt_lo);
  k_conv<<<2048, 128, 0, stream>>>(xt_hi, xt_lo, wb_hi, wb_lo, zbuf, pbuf, out);
  k_topk<<<1024, 128, 0, stream>>>(pbuf, x, winH, winW, winX, winO);
  k_pairs<<<BB, 256, 0, stream>>>(winH, winW, winX, winO, Aacc, Ssum, Osum);
  k_final<<<(CC * CC * PS * PS + 255) / 256, 256, 0, stream>>>(W, Aacc, Ssum, Osum, newW);
}
